// KAN_Layer_15350213116057
// MI455X (gfx1250) — matrix operand order repaired and checked by reading, not compiled, not run on hardware
//
#include <hip/hip_runtime.h>
#include <hip/hip_bf16.h>

// KAN layer as structured-sparse GEMM on WMMA:
//   out[2048,256] = W[2048,4096] @ coeffs^T[4096,256]
//   W[b, i*32+k] = max(0, 1 - |k - pos(b,i)|),  pos = (x+1)*15.5 clamped to [0,31]
// W tiles are synthesized in registers (zero gathers), coeffs staged via LDS,
// accumulation in f32 via v_wmma_f32_16x16x32_f16 (K=32 == NK).
// v2: 8 o-tiles per wave so one A-fragment build feeds 8 WMMAs (halves VALU:WMMA).

#define B_TOTAL 2048
#define IN_F    128
#define OUT_F   256
#define NK      32
#define ICHUNK  4
#define NCHUNK  (IN_F / ICHUNK)   // 32
#define OTILES  8                 // 16-wide o-tiles per wave -> WG covers 128 o's

typedef _Float16 v16h __attribute__((ext_vector_type(16)));
typedef _Float16 h8   __attribute__((ext_vector_type(8)));
typedef float    v8f  __attribute__((ext_vector_type(8)));

__global__ __launch_bounds__(256) void kan_wmma_kernel(
    const float* __restrict__ x,       // [2048,128]
    const float* __restrict__ coeffs,  // [256,128,32]
    float* __restrict__ out)           // [2048,256]
{
    // LDS staging: coeffs chunk (f16) + x chunk (f32)
    __shared__ __align__(16) _Float16 sB[ICHUNK][128][NK];  // 32 KB
    __shared__ __align__(16) float    sX[128][ICHUNK];      //  2 KB

    const int tid    = threadIdx.x;
    const int lane   = tid & 31;
    const int wave   = tid >> 5;           // 0..7 -> b sub-tile
    const int m      = lane & 15;          // row/col within 16-wide tile
    const int hi     = (lane & 16) >> 4;   // lane half (0/1)
    const int b_base = blockIdx.x * 128;   // 16 b-groups
    const int o_base = blockIdx.y * 128;   // 2  o-groups

    const int k0     = hi << 4;            // B K-half: 0 or 16
    const float kbA  = (float)(hi << 4);   // A K-base offset: 0 or 16

    v8f acc[OTILES];
    const v8f vzero = {0.f,0.f,0.f,0.f,0.f,0.f,0.f,0.f};
#pragma unroll
    for (int ot = 0; ot < OTILES; ++ot) acc[ot] = vzero;

    for (int chunk = 0; chunk < NCHUNK; ++chunk) {
        const int i0 = chunk * ICHUNK;

        // ---- stage x[b_base..+128, i0..+4] -> sX (one float4 per b-row) ----
        if (tid < 128) {
            float4 v = *reinterpret_cast<const float4*>(
                x + (size_t)(b_base + tid) * IN_F + i0);
            *reinterpret_cast<float4*>(&sX[tid][0]) = v;
        }

        // ---- stage coeffs[o_base..+128, i0..+4, 0..32] -> sB (f32 -> f16) ----
#pragma unroll
        for (int c = tid; c < 512; c += 256) {
            const int ii = c >> 7;          // 0..3
            const int o  = c & 127;         // 0..127
            const float4* s4 = reinterpret_cast<const float4*>(
                coeffs + ((size_t)(o_base + o) * IN_F + (i0 + ii)) * NK);
            h8* dst = reinterpret_cast<h8*>(&sB[ii][o][0]);
#pragma unroll
            for (int q = 0; q < 4; ++q) {
                float4 lo = s4[2 * q];
                float4 hv4 = s4[2 * q + 1];
                h8 hv;
                hv[0] = (_Float16)lo.x;  hv[1] = (_Float16)lo.y;
                hv[2] = (_Float16)lo.z;  hv[3] = (_Float16)lo.w;
                hv[4] = (_Float16)hv4.x; hv[5] = (_Float16)hv4.y;
                hv[6] = (_Float16)hv4.z; hv[7] = (_Float16)hv4.w;
                dst[q] = hv;
            }
        }

        // ---- prefetch part of next chunk's coeffs (global_prefetch_b8) ----
        if (chunk + 1 < NCHUNK) {
            const int ii = tid >> 7;        // 0..1 (cacheline granularity covers rest)
            const int o  = tid & 127;
            __builtin_prefetch(
                coeffs + ((size_t)(o_base + o) * IN_F + (i0 + ICHUNK + ii)) * NK, 0, 0);
        }

        __syncthreads();

        // per-lane x values for this chunk (one ds_load_b128)
        const float4 xq = *reinterpret_cast<const float4*>(&sX[wave * 16 + m][0]);
        const float xv[ICHUNK] = {xq.x, xq.y, xq.z, xq.w};

#pragma unroll
        for (int ii = 0; ii < ICHUNK; ++ii) {
            // normalized spline position for (b_row, i0+ii)
            float pos = (xv[ii] + 1.0f) * 15.5f;
            pos = fminf(fmaxf(pos, 0.0f), 31.0f);

            // Synthesize A fragment: hat-function weights == linear interp.
            // A 16x32 layout (matches B): K_j = j + (lane[4] ? 16 : 0)
            v16h a;
#pragma unroll
            for (int j = 0; j < 16; ++j) {
                const float kk = (float)j + kbA;
                float w = 1.0f - fabsf(kk - pos);
                a[j] = (_Float16)fmaxf(w, 0.0f);
            }

            // 8 o-tiles share this A fragment
#pragma unroll
            for (int ot = 0; ot < OTILES; ++ot) {
                const int o_l = ot * 16 + m;   // B layout: N = lane&15, K-half by lane[4]
                v16h bv = *reinterpret_cast<const v16h*>(&sB[ii][o_l][k0]);
                acc[ot] = __builtin_amdgcn_wmma_f32_16x16x32_f16(
                    false, a, false, bv, (short)0, acc[ot], false, false);
            }
        }

        __syncthreads();   // protect sB/sX before next chunk's staging
    }

    // ---- epilogue: C/D layout -> out. VGPR r: M = r + 8*lane[4], N = lane&15 ----
    const int M0 = (lane & 16) >> 1;
#pragma unroll
    for (int ot = 0; ot < OTILES; ++ot) {
#pragma unroll
        for (int r = 0; r < 8; ++r) {
            const int M = r + M0;
            out[(size_t)(b_base + wave * 16 + M) * OUT_F + (o_base + ot * 16 + m)]
                = acc[ot][r];
        }
    }
}

extern "C" void kernel_launch(void* const* d_in, const int* in_sizes, int n_in,
                              void* d_out, int out_size, void* d_ws, size_t ws_size,
                              hipStream_t stream) {
    const float* x      = (const float*)d_in[0];   // [2048,128] f32
    const float* coeffs = (const float*)d_in[1];   // [256,128,32] f32
    float* out          = (float*)d_out;           // [2048,256] f32
    (void)in_sizes; (void)n_in; (void)out_size; (void)d_ws; (void)ws_size;

    dim3 grid(B_TOTAL / 128, OUT_F / 128);         // 16 x 2 workgroups
    kan_wmma_kernel<<<grid, 256, 0, stream>>>(x, coeffs, out);
}
